// InfoNCELoss_34540126994859
// MI455X (gfx1250) — compile-verified, matrix operand order checked by reading, not run on hardware
//
#include <hip/hip_runtime.h>
#include <hip/hip_bf16.h>

typedef __attribute__((ext_vector_type(8)))  _Float16 v8h;
typedef __attribute__((ext_vector_type(16))) _Float16 v16h;
typedef __attribute__((ext_vector_type(8)))  float    v8f;

#define NROWS 8192
#define DDIM  256
#define CSPLIT 4
#define JT_PER_CS (NROWS / 16 / CSPLIT)   // 128 column tiles of 16 per chunk

#define ROW_BYTES   (DDIM * 2)            // 512 B per f16 row
#define LDS_STRIDE  (ROW_BYTES + 16)      // 528 B: b128 reads stripe all 64 banks
#define TILE_BYTES  (16 * LDS_STRIDE)     // 8448 B per staged B tile
#define NBUF        4                     // async pipeline depth 2, 4 buffers

// exp2 / log2 fast paths (v_exp_f32 / v_log_f32); arg range is bounded (|x|<=2.89)
#if __has_builtin(__builtin_amdgcn_exp2f)
#define EXP2F(x) __builtin_amdgcn_exp2f(x)
#else
#define EXP2F(x) exp2f(x)
#endif
#if __has_builtin(__builtin_amdgcn_logf)
#define LOG2F(x) __builtin_amdgcn_logf(x)
#else
#define LOG2F(x) log2f(x)
#endif
#if __has_builtin(__builtin_amdgcn_sched_barrier)
#define SCHED_FENCE() __builtin_amdgcn_sched_barrier(0)
#else
#define SCHED_FENCE()
#endif

#define TWO_LOG2E 2.8853900817779268f   // 2 * log2(e)  (sim = dot / 0.5)
#define LN2       0.6931471805599453f

// ---------------------------------------------------------------------------
// 1) L2-normalize rows, f32 -> f16
// ---------------------------------------------------------------------------
__global__ __launch_bounds__(256) void normalize_kernel(const float* __restrict__ in,
                                                        _Float16* __restrict__ out) {
  __shared__ float red[256];
  const int row = blockIdx.x;
  const int tid = threadIdx.x;
  const float x = in[(size_t)row * DDIM + tid];
  red[tid] = x * x;
  __syncthreads();
#pragma unroll
  for (int s = 128; s > 0; s >>= 1) {
    if (tid < s) red[tid] += red[tid + s];
    __syncthreads();
  }
  const float norm  = sqrtf(red[0]);
  const float scale = 1.0f / fmaxf(norm, 1e-12f);
  out[(size_t)row * DDIM + tid] = (_Float16)(x * scale);
}

// ---------------------------------------------------------------------------
// 2) diagonal: sim[i,i] = 2 * dot(qn_i, kn_i)   (one wave32 per row)
// ---------------------------------------------------------------------------
__global__ __launch_bounds__(256) void diag_kernel(const _Float16* __restrict__ qn,
                                                   const _Float16* __restrict__ kn,
                                                   float* __restrict__ diag) {
  const int wave = threadIdx.x >> 5;
  const int lane = threadIdx.x & 31;
  const int row  = blockIdx.x * 8 + wave;
  const v8h qv = *(const v8h*)(qn + (size_t)row * DDIM + lane * 8);
  const v8h kv = *(const v8h*)(kn + (size_t)row * DDIM + lane * 8);
  float dot = 0.f;
#pragma unroll
  for (int i = 0; i < 8; ++i) dot += (float)qv[i] * (float)kv[i];
#pragma unroll
  for (int off = 1; off < 32; off <<= 1) dot += __shfl_xor(dot, off, 32);
  if (lane == 0) diag[row] = 2.0f * dot;
}

// ---------------------------------------------------------------------------
// Async prefetch of one 16-row B tile (8 KB contiguous) into LDS, cooperative
// across all 8 waves: each wave issues 2x GLOBAL_LOAD_ASYNC_TO_LDS_B128
// (32 lanes x 16 B = 512 B per instruction). Tracked by ASYNCcnt.
// LDS destination is padded: row r -> lds_buf + r*528 + col.
// ---------------------------------------------------------------------------
__device__ __forceinline__ void prefetch_btile(unsigned long long gbase,
                                               unsigned int lds_buf,
                                               int wave, int lane) {
  const int g0 = wave * 1024 + lane * 16;   // byte offset in 8 KB tile, instr 0
  const int g1 = g0 + 512;                  // instr 1
  const unsigned int l0 = lds_buf + (unsigned int)((g0 >> 9) * LDS_STRIDE + (g0 & 511));
  const unsigned int l1 = lds_buf + (unsigned int)((g1 >> 9) * LDS_STRIDE + (g1 & 511));
  asm volatile("global_load_async_to_lds_b128 %0, %1, %4\n\t"
               "global_load_async_to_lds_b128 %2, %3, %4"
               :
               : "v"(l0), "v"(g0), "v"(l1), "v"(g1), "s"(gbase)
               : "memory");
}

// ---------------------------------------------------------------------------
// 3) fused GEMM + sum(exp(sim)) per row.
//    Wave owns 16 rows; A tile (16 x 256 f16) resident in 64 VGPRs.
//    B tiles staged in LDS via async copies, pipeline depth 2 (4 buffers),
//    ONE barrier per tile. Per 16-col tile: preload all 8 B fragments
//    (sched_barrier pins the ds_load clause), then 8x chained WMMA,
//    then v_exp + accumulate.
// ---------------------------------------------------------------------------
__global__ __launch_bounds__(256) void gemm_lse_kernel(const _Float16* __restrict__ qn,
                                                       const _Float16* __restrict__ kn,
                                                       float* __restrict__ partial) {
  __shared__ __align__(16) char bsm[NBUF * TILE_BYTES];

  const int wave = threadIdx.x >> 5;
  const int lane = threadIdx.x & 31;
  const int rb   = blockIdx.x / CSPLIT;      // row-block 0..63
  const int cs   = blockIdx.x % CSPLIT;      // column chunk 0..3
  const int m0   = rb * 128 + wave * 16;     // first of this wave's 16 rows
  const int mrow = m0 + (lane & 15);         // A: lane holds row m0 + lane%16
  const int half = lane >> 4;                // K sub-pattern per lane half

  // ---- load resident A tile: a[t] covers K = [32t, 32t+32) -----------------
  // 16-bit A 16x32 layout: lane half 0 -> K = {0..7, 16..23}, half 1 -> {8..15, 24..31}
  const _Float16* abase = qn + (size_t)mrow * DDIM + half * 8;
  v16h a[8];
#pragma unroll
  for (int t = 0; t < 8; ++t) {
    v8h lo = *(const v8h*)(abase + t * 32);
    v8h hi = *(const v8h*)(abase + t * 32 + 16);
    a[t] = __builtin_shufflevector(lo, hi, 0, 1, 2, 3, 4, 5, 6, 7,
                                           8, 9, 10, 11, 12, 13, 14, 15);
  }

  float ssum[8];
#pragma unroll
  for (int v = 0; v < 8; ++v) ssum[v] = 0.f;

  const int j0 = cs * JT_PER_CS;
  const int j1 = j0 + JT_PER_CS;
  const unsigned int smbase = (unsigned int)(uintptr_t)&bsm[0]; // LDS byte offset
  const unsigned long long knbase = (unsigned long long)(uintptr_t)kn;

  // prime the pipeline: tiles j0, j0+1 -> buffers 0, 1
  prefetch_btile(knbase + (unsigned long long)j0 * (16 * ROW_BYTES),
                 smbase, wave, lane);
  prefetch_btile(knbase + (unsigned long long)(j0 + 1) * (16 * ROW_BYTES),
                 smbase + TILE_BYTES, wave, lane);

  for (int j = j0; j < j1; ++j) {
    const int cur = (j - j0) & (NBUF - 1);
    // Issue tile j+2 into buffer (cur+2)%4. Safe with a single barrier per
    // iteration: passing barrier(j-1) implies ALL waves finished computing tile
    // j-2, which is the last reader of that buffer.
    if (j + 2 < j1) {
      prefetch_btile(knbase + (unsigned long long)(j + 2) * (16 * ROW_BYTES),
                     smbase + (unsigned int)(((cur + 2) & (NBUF - 1)) * TILE_BYTES),
                     wave, lane);
      asm volatile("s_wait_asynccnt 0x4" ::: "memory"); // tile j done; j+1,j+2 in flight
    } else if (j + 1 < j1) {
      asm volatile("s_wait_asynccnt 0x2" ::: "memory"); // tile j done; j+1 in flight
    } else {
      asm volatile("s_wait_asynccnt 0x0" ::: "memory");
    }
    __syncthreads();   // every wave's share of tile j is in LDS

    // B = kn^T tile from LDS: lane holds column (= kn row) j*16 + lane%16.
    // Preload ALL fragments, then fence the scheduler so the 16 ds_load_b128
    // stay clustered ahead of the WMMA chain (staggered s_wait_dscnt).
    const char* bbuf = bsm + cur * TILE_BYTES + (lane & 15) * LDS_STRIDE + half * 16;
    v16h b[8];
#pragma unroll
    for (int t = 0; t < 8; ++t) {
      v8h lo = *(const v8h*)(bbuf + t * 64);        // K = 32t + half*8 ..
      v8h hi = *(const v8h*)(bbuf + t * 64 + 32);   // K = 32t + 16 + half*8 ..
      b[t] = __builtin_shufflevector(lo, hi, 0, 1, 2, 3, 4, 5, 6, 7,
                                             8, 9, 10, 11, 12, 13, 14, 15);
    }
    SCHED_FENCE();
    v8f c = {};
#pragma unroll
    for (int t = 0; t < 8; ++t) {
      c = __builtin_amdgcn_wmma_f32_16x16x32_f16(false, a[t], false, b[t],
                                                 (short)0, c, false, false);
    }
    // c[v] = dot for row (m0 + v + 8*half), col (j*16 + lane%16); bounded [-1,1]
#pragma unroll
    for (int v = 0; v < 8; ++v) ssum[v] += EXP2F(c[v] * TWO_LOG2E);
  }

  // reduce the 16 columns held across each 16-lane half
#pragma unroll
  for (int v = 0; v < 8; ++v) {
    float s = ssum[v];
    s += __shfl_xor(s, 1, 32);
    s += __shfl_xor(s, 2, 32);
    s += __shfl_xor(s, 4, 32);
    s += __shfl_xor(s, 8, 32);
    ssum[v] = s;
  }
  // lane 0 -> rows m0..m0+7, lane 16 -> rows m0+8..m0+15 (each written once: no atomics)
  if (lane == 0) {
#pragma unroll
    for (int v = 0; v < 8; ++v) partial[(size_t)cs * NROWS + m0 + v] = ssum[v];
  } else if (lane == 16) {
#pragma unroll
    for (int v = 0; v < 8; ++v) partial[(size_t)cs * NROWS + m0 + 8 + v] = ssum[v];
  }
}

// ---------------------------------------------------------------------------
// 4) loss_i = log(S_i) - diag_i ; mean over rows (fixed-order => deterministic)
// ---------------------------------------------------------------------------
__global__ __launch_bounds__(1024) void finalize_kernel(const float* __restrict__ partial,
                                                        const float* __restrict__ diag,
                                                        float* __restrict__ out) {
  __shared__ float red[1024];
  const int tid = threadIdx.x;
  float acc = 0.f;
  for (int r = tid; r < NROWS; r += 1024) {
    float S = 0.f;
#pragma unroll
    for (int c = 0; c < CSPLIT; ++c) S += partial[(size_t)c * NROWS + r];
    acc += LOG2F(S) * LN2 - diag[r];
  }
  red[tid] = acc;
  __syncthreads();
#pragma unroll
  for (int s = 512; s > 0; s >>= 1) {
    if (tid < s) red[tid] += red[tid + s];
    __syncthreads();
  }
  if (tid == 0) out[0] = red[0] / (float)NROWS;
}

// ---------------------------------------------------------------------------
extern "C" void kernel_launch(void* const* d_in, const int* in_sizes, int n_in,
                              void* d_out, int out_size, void* d_ws, size_t ws_size,
                              hipStream_t stream) {
  (void)in_sizes; (void)n_in; (void)out_size; (void)ws_size;
  const float* q = (const float*)d_in[0];
  const float* k = (const float*)d_in[1];
  float* out = (float*)d_out;

  char* ws = (char*)d_ws;
  _Float16* qn   = (_Float16*)(ws);                       // 4 MB
  _Float16* kn   = (_Float16*)(ws + 4194304);             // 4 MB
  float* diag    = (float*)(ws + 8388608);                // 32 KB
  float* partial = (float*)(ws + 8388608 + 32768);        // 128 KB

  normalize_kernel<<<NROWS, 256, 0, stream>>>(q, qn);
  normalize_kernel<<<NROWS, 256, 0, stream>>>(k, kn);
  diag_kernel<<<NROWS / 8, 256, 0, stream>>>(qn, kn, diag);
  gemm_lse_kernel<<<(NROWS / 128) * CSPLIT, 256, 0, stream>>>(qn, kn, partial);
  finalize_kernel<<<1, 1024, 0, stream>>>(partial, diag, out);
}
